// GATAttn_32427003085396
// MI455X (gfx1250) — matrix operand order repaired and checked by reading, not compiled, not run on hardware
//
#include <hip/hip_runtime.h>
#include <hip/hip_bf16.h>

// ---------------------------------------------------------------------------
// GAT attention layer, fused flash-attention style for MI455X (gfx1250).
//   N=8192, F_IN=512, F_OUT=64.  HBM-bound on the 256MB adj matrix (~11us at
//   23.3 TB/s); matrix math via v_wmma_f32_16x16x32_bf16 (wave32).
//   bf16 packing via v_perm_b32; online-softmax rescale skipped via ballot;
//   W pre-converted to bf16 once so the GEMM B-fragments load directly.
// ---------------------------------------------------------------------------

typedef __attribute__((ext_vector_type(16))) __bf16        v16bf;
typedef __attribute__((ext_vector_type(8)))  float         v8f;
typedef __attribute__((ext_vector_type(8)))  unsigned int  v8u;

#define NN     8192
#define FIN    512
#define FOUT   64
#define SPLITS 4
#define LEAKY  0.2f

__device__ __forceinline__ unsigned short f2bfbits(float f) {
  unsigned u = __builtin_bit_cast(unsigned, f);
  u += 0x7FFFu + ((u >> 16) & 1u);          // round-to-nearest-even
  return (unsigned short)(u >> 16);
}
// Pack two floats into one dword of two bf16 (round-to-nearest via +0x8000,
// then a single v_perm_b32 grabs the two high halves).
__device__ __forceinline__ unsigned packbf(float a, float b) {
  unsigned ua = __builtin_bit_cast(unsigned, a) + 0x8000u;
  unsigned ub = __builtin_bit_cast(unsigned, b) + 0x8000u;
  return __builtin_amdgcn_perm(ub, ua, 0x07060302u);  // {b.hi16, a.hi16}
}
__device__ __forceinline__ v16bf pack16(const float* f) {
  v8u u;
  #pragma unroll
  for (int q = 0; q < 8; ++q) u[q] = packbf(f[2 * q], f[2 * q + 1]);
  return __builtin_bit_cast(v16bf, u);
}
__device__ __forceinline__ void load8f(const float* p, float* d) {
  float4 v0 = *(const float4*)(p);
  float4 v1 = *(const float4*)(p + 4);
  d[0] = v0.x; d[1] = v0.y; d[2] = v0.z; d[3] = v0.w;
  d[4] = v1.x; d[5] = v1.y; d[6] = v1.z; d[7] = v1.w;
}
__device__ __forceinline__ void load8i(const int* p, int* d) {
  int4 v0 = *(const int4*)(p);
  int4 v1 = *(const int4*)(p + 4);
  d[0] = v0.x; d[1] = v0.y; d[2] = v0.z; d[3] = v0.w;
  d[4] = v1.x; d[5] = v1.y; d[6] = v1.z; d[7] = v1.w;
}

// ---------------------------------------------------------------------------
// Kernel 0: one-shot W (fp32 [512,64]) -> bf16, row-major.  32768 elements.
// ---------------------------------------------------------------------------
__global__ __launch_bounds__(256) void wbf_prep_kernel(
    const float* __restrict__ W, unsigned short* __restrict__ wbf)
{
  const int idx = (blockIdx.x * 256 + threadIdx.x) * 2;
  wbf[idx]     = (unsigned short)(packbf(W[idx], W[idx + 1]) & 0xFFFFu);
  wbf[idx + 1] = (unsigned short)(packbf(W[idx], W[idx + 1]) >> 16);
}

// ---------------------------------------------------------------------------
// Kernel 1: Wh = x @ W  (bf16 WMMA, fp32 accumulate).  One wave per 16-row
// tile; 4 waves per block.  Also stores a bf16 copy of Wh for the P@V WMMA.
// A-frag (16-bit, 16x32): lane L<16 -> row L, K={0..7,16..23}; lane L+16 ->
// row L, K={8..15,24..31}.  B-frag: lane -> column N=L%16, same K order as A,
// gathered from the pre-converted bf16 W.
// ---------------------------------------------------------------------------
__global__ __launch_bounds__(128) void wh_gemm_kernel(
    const float* __restrict__ x, const unsigned short* __restrict__ wbf,
    float* __restrict__ wh, unsigned short* __restrict__ whbf)
{
  const int lane = threadIdx.x & 31;
  const int wave = threadIdx.x >> 5;
  const int tile = blockIdx.x * 4 + wave;     // 0..511
  const int i0   = tile * 16;
  const int row  = lane & 15;
  const int hi   = lane >> 4;
  const int koff = hi * 8;

  const v8f zero = {0.f, 0.f, 0.f, 0.f, 0.f, 0.f, 0.f, 0.f};
  v8f c[4];
  #pragma unroll
  for (int t = 0; t < 4; ++t) c[t] = zero;

  const float* xrow = x + (size_t)(i0 + row) * FIN;

  for (int k = 0; k < FIN; k += 32) {
    float av[16];
    load8f(xrow + k + koff,      av);
    load8f(xrow + k + koff + 16, av + 8);
    const v16bf af = pack16(av);

    const unsigned short* wrow = wbf + (size_t)(k + koff) * FOUT + (lane & 15);
    #pragma unroll
    for (int t = 0; t < 4; ++t) {
      v16bf bfr;
      #pragma unroll
      for (int q = 0; q < 16; ++q)
        bfr[q] = __builtin_bit_cast(__bf16, wrow[(size_t)(q < 8 ? q : q + 8) * FOUT + t * 16]);
      c[t] = __builtin_amdgcn_wmma_f32_16x16x32_bf16(
          false, af, false, bfr, (short)0, c[t], false, false);
    }
  }

  // C layout: lanes 0-15 -> N=lane, rows 0..7 in vgprs; lanes 16-31 -> rows 8..15
  #pragma unroll
  for (int r = 0; r < 8; ++r) {
    const int orow = i0 + r + 8 * hi;
    #pragma unroll
    for (int t = 0; t < 4; ++t) {
      const int col = t * 16 + (lane & 15);
      float v = c[t][r];
      wh[(size_t)orow * FOUT + col]   = v;
      whbf[(size_t)orow * FOUT + col] = f2bfbits(v);
    }
  }
}

// ---------------------------------------------------------------------------
// Kernel 2: wh_i = Wh @ a[:64], wh_j = Wh @ a[64:].  One thread per row.
// ---------------------------------------------------------------------------
__global__ __launch_bounds__(256) void rowdot_kernel(
    const float* __restrict__ wh, const float* __restrict__ a,
    float* __restrict__ wh_i, float* __restrict__ wh_j)
{
  const int r = blockIdx.x * blockDim.x + threadIdx.x;
  const float* p = wh + (size_t)r * FOUT;
  float si = 0.f, sj = 0.f;
  #pragma unroll
  for (int c = 0; c < FOUT; ++c) {
    float v = p[c];
    si += v * a[c];
    sj += v * a[FOUT + c];
  }
  wh_i[r] = si;
  wh_j[r] = sj;
}

// ---------------------------------------------------------------------------
// Kernel 3: fused masked-softmax attention (online softmax, single pass over
// adj).  One wave per (16-row i-tile, j-split).  Per 32-wide j step:
//   e = leaky(wh_i + wh_j) masked by adj  -> online max/sum -> p = exp(e-m)
//   O (16x64, fp32 C-frags) += P(bf16) @ Wh_bf(bf16) via 4 WMMAs.
// Rescale only runs when some row's max grew (wave-uniform ballot guard).
// ---------------------------------------------------------------------------
__global__ __launch_bounds__(128) void attn_kernel(
    const int* __restrict__ adj, const unsigned short* __restrict__ whbf,
    const float* __restrict__ wh_i, const float* __restrict__ wh_j,
    float* __restrict__ m_part, float* __restrict__ l_part,
    float* __restrict__ o_part)
{
  const int lane = threadIdx.x & 31;
  const int wave = threadIdx.x >> 5;
  const int tile = blockIdx.x * 4 + wave;     // 0..511
  const int s    = blockIdx.y;                // 0..SPLITS-1
  const int i0   = tile * 16;
  const int row  = lane & 15;
  const int hi   = lane >> 4;
  const int koff = hi * 8;

  const float NEG_INF = -__builtin_inff();
  const float whi = wh_i[i0 + row];

  float m = -1.0e30f;      // running row max (row = lane&15)
  float l = 0.f;           // running row sum
  const v8f zero = {0.f, 0.f, 0.f, 0.f, 0.f, 0.f, 0.f, 0.f};
  v8f o[4];
  #pragma unroll
  for (int t = 0; t < 4; ++t) o[t] = zero;

  const size_t arow = (size_t)(i0 + row) * NN;
  const int jbeg = s * (NN / SPLITS);
  const int jend = jbeg + (NN / SPLITS);

  #pragma unroll 2
  for (int j0 = jbeg; j0 < jend; j0 += 32) {
    if (j0 + 64 < jend)
      __builtin_prefetch(adj + arow + j0 + 64 + koff, 0, 1);

    // 16 j's per lane, ordered to match the 16-bit A-frag layout:
    // element q -> j = j0 + koff + (q<8 ? q : q+8)
    int   av[16];
    float wv[16];
    load8i(adj + arow + j0 + koff,      av);
    load8i(adj + arow + j0 + koff + 16, av + 8);
    load8f(wh_j + j0 + koff,            wv);
    load8f(wh_j + j0 + koff + 16,       wv + 8);

    float e[16];
    float cmax = -1.0e30f;
    #pragma unroll
    for (int q = 0; q < 16; ++q) {
      float ev = whi + wv[q];
      ev = fmaxf(ev, LEAKY * ev);                // leaky relu (alpha < 1)
      ev = av[q] > 0 ? ev : NEG_INF;             // adjacency mask
      e[q] = ev;
      cmax = fmaxf(cmax, ev);
    }
    cmax = fmaxf(cmax, __shfl_xor(cmax, 16, 32));   // combine both K-halves

    // Online max update; skip entirely when no row grew (wave-uniform).
    if (__ballot(cmax > m)) {
      const float mnew  = fmaxf(m, cmax);
      const float scale = __expf(m - mnew);
      m = mnew;
      l *= scale;
      // vgpr r of a lane holds row (r + 8*hi); scale for row R is in lane R.
      #pragma unroll
      for (int r = 0; r < 8; ++r) {
        float fr = __shfl(scale, r + 8 * hi, 32);
        #pragma unroll
        for (int t = 0; t < 4; ++t) o[t][r] = o[t][r] * fr;
      }
    }

    float p[16];
    float psum = 0.f;
    #pragma unroll
    for (int q = 0; q < 16; ++q) {
      p[q] = __expf(e[q] - m);                    // exp(-inf)=0 for masked
      psum += p[q];
    }
    psum += __shfl_xor(psum, 16, 32);
    l += psum;
    const v16bf pa = pack16(p);

    // V fragments: lane -> column N=lane%16, K order matching the A-frag.
    const unsigned short* vrow = whbf + (size_t)(j0 + koff) * FOUT + (lane & 15);
    #pragma unroll
    for (int t = 0; t < 4; ++t) {
      v16bf bfr;
      #pragma unroll
      for (int q = 0; q < 16; ++q)
        bfr[q] = __builtin_bit_cast(__bf16, vrow[(size_t)(q < 8 ? q : q + 8) * FOUT + t * 16]);
      o[t] = __builtin_amdgcn_wmma_f32_16x16x32_bf16(
          false, pa, false, bfr, (short)0, o[t], false, false);
    }
  }

  // Write partials.
  float* obuf = o_part + ((size_t)tile * SPLITS + s) * (16 * FOUT);
  #pragma unroll
  for (int r = 0; r < 8; ++r) {
    const int orow = r + 8 * hi;
    #pragma unroll
    for (int t = 0; t < 4; ++t)
      obuf[orow * FOUT + t * 16 + (lane & 15)] = o[t][r];
  }
  if (hi == 0) {
    m_part[((size_t)tile * SPLITS + s) * 16 + row] = m;
    l_part[((size_t)tile * SPLITS + s) * 16 + row] = l;
  }
}

// ---------------------------------------------------------------------------
// Kernel 4: merge the j-splits with softmax rescaling, normalize, ELU, store.
// One block per i-tile; thread -> (row, 4 cols).
// ---------------------------------------------------------------------------
__global__ __launch_bounds__(256) void combine_kernel(
    const float* __restrict__ m_part, const float* __restrict__ l_part,
    const float* __restrict__ o_part, float* __restrict__ out)
{
  const int tile = blockIdx.x;            // 0..511
  const int tid  = threadIdx.x;           // 0..255
  const int r    = tid >> 4;              // row 0..15
  const int c0   = (tid & 15) * 4;        // col base

  float ms[SPLITS], sw[SPLITS];
  float mmax = -1.0e30f;
  #pragma unroll
  for (int s = 0; s < SPLITS; ++s) {
    ms[s] = m_part[((size_t)tile * SPLITS + s) * 16 + r];
    mmax = fmaxf(mmax, ms[s]);
  }
  float denom = 0.f;
  #pragma unroll
  for (int s = 0; s < SPLITS; ++s) {
    sw[s] = __expf(ms[s] - mmax);
    denom += sw[s] * l_part[((size_t)tile * SPLITS + s) * 16 + r];
  }
  const float inv = 1.0f / denom;

  #pragma unroll
  for (int cc = 0; cc < 4; ++cc) {
    const int c = c0 + cc;
    float acc = 0.f;
    #pragma unroll
    for (int s = 0; s < SPLITS; ++s)
      acc += sw[s] * o_part[((size_t)tile * SPLITS + s) * (16 * FOUT) + r * FOUT + c];
    float h = acc * inv;
    h = h > 0.f ? h : (__expf(h) - 1.f);        // ELU (alpha = 1)
    out[(size_t)(tile * 16 + r) * FOUT + c] = h;
  }
}

// ---------------------------------------------------------------------------
extern "C" void kernel_launch(void* const* d_in, const int* in_sizes, int n_in,
                              void* d_out, int out_size, void* d_ws, size_t ws_size,
                              hipStream_t stream)
{
  const float* x   = (const float*)d_in[0];   // [8192, 512]
  const int*   adj = (const int*)d_in[1];     // [8192, 8192]
  const float* W   = (const float*)d_in[2];   // [512, 64]
  const float* a   = (const float*)d_in[3];   // [128, 1]
  float* out = (float*)d_out;                 // [8192, 64]

  char* ws = (char*)d_ws;
  size_t off = 0;
  auto alloc = [&](size_t bytes) -> void* {
    void* p = ws + off;
    off += (bytes + 255) & ~(size_t)255;
    return p;
  };
  float*          wh     = (float*)alloc((size_t)NN * FOUT * sizeof(float));
  unsigned short* whbf   = (unsigned short*)alloc((size_t)NN * FOUT * sizeof(unsigned short));
  unsigned short* wbf    = (unsigned short*)alloc((size_t)FIN * FOUT * sizeof(unsigned short));
  float*          wh_i   = (float*)alloc((size_t)NN * sizeof(float));
  float*          wh_j   = (float*)alloc((size_t)NN * sizeof(float));
  float*          m_part = (float*)alloc((size_t)(NN / 16) * SPLITS * 16 * sizeof(float));
  float*          l_part = (float*)alloc((size_t)(NN / 16) * SPLITS * 16 * sizeof(float));
  float*          o_part = (float*)alloc((size_t)(NN / 16) * SPLITS * 16 * FOUT * sizeof(float));

  wbf_prep_kernel<<<dim3((FIN * FOUT) / 512), dim3(256), 0, stream>>>(W, wbf);
  wh_gemm_kernel<<<dim3(NN / 64), dim3(128), 0, stream>>>(x, wbf, wh, whbf);
  rowdot_kernel<<<dim3(NN / 256), dim3(256), 0, stream>>>(wh, a, wh_i, wh_j);
  attn_kernel<<<dim3(NN / 64, SPLITS), dim3(128), 0, stream>>>(
      adj, whbf, wh_i, wh_j, m_part, l_part, o_part);
  combine_kernel<<<dim3(NN / 16), dim3(256), 0, stream>>>(m_part, l_part, o_part, out);
}
